// SentimentLSTM_90323162235598
// MI455X (gfx1250) — matrix operand order repaired and checked by reading, not compiled, not run on hardware
//
#include <hip/hip_runtime.h>
#include <hip/hip_bf16.h>

// ---------------------------------------------------------------------------
// Types for CDNA5 WMMA (wave32): D(16x16 f32) = A(16x32 bf16) x B(32x16 bf16) + C
// ---------------------------------------------------------------------------
typedef __attribute__((ext_vector_type(16))) __bf16 v16bf;
typedef __attribute__((ext_vector_type(8)))  __bf16 bf16x8;
typedef __attribute__((ext_vector_type(8)))  float  v8f;

typedef __attribute__((ext_vector_type(4))) unsigned int tdm_u32x4;
typedef __attribute__((ext_vector_type(8))) int          tdm_i32x8;
typedef __attribute__((ext_vector_type(4))) int          tdm_i32x4;

#define BATCH 64
#define SEQ   512
#define EMB   512
#define HID   1024
#define G4    4096
#define ROWS  (SEQ * BATCH)   // 32768 time-major rows

// Blocked LDS layout for WMMA operands: [k-block of 32][row][32 elems],
// with a 16B (8-elem) pad after every 4 rows -> bank-conflict-free frag loads.
//   elem_off(row,k) = row*32 + (row>>2)*8 + (k&31)
//   block stride    = 64*32 + 16*8 = 2176 elems (4352 B, 16B-aligned)
#define BLK_ROW(r)   ((r) * 32 + ((r) >> 2) * 8)
#define BLK_STRIDE64 2176                    // elems per 64-row k-block

__device__ __forceinline__ v8f wmma_bf16(v16bf a, v16bf b, v8f c) {
  return __builtin_amdgcn_wmma_f32_16x16x32_bf16(false, a, false, b, (short)0, c,
                                                 false, false);
}

// A fragment from a blocked 32-k tile: rows row0..row0+15, K 0..31 of block.
__device__ __forceinline__ v16bf load_frag_a_blk(const __bf16* blk, int row0, int lane) {
  int r  = row0 + (lane & 15);
  int kb = (lane >> 4) * 8;
  const __bf16* p = blk + BLK_ROW(r) + kb;
  bf16x8 lo = *(const bf16x8*)(p);
  bf16x8 hi = *(const bf16x8*)(p + 16);
  v16bf out;
#pragma unroll
  for (int i = 0; i < 8; ++i) { out[i] = lo[i]; out[i + 8] = hi[i]; }
  return out;
}

// B fragment (B stored transposed: "row" = output column, 32 k contiguous).
__device__ __forceinline__ v16bf load_frag_b_blk(const __bf16* blk, int col0, int lane) {
  int c  = col0 + (lane & 15);
  int kb = (lane >> 4) * 8;
  const __bf16* p = blk + BLK_ROW(c) + kb;
  bf16x8 lo = *(const bf16x8*)(p);
  bf16x8 hi = *(const bf16x8*)(p + 16);
  v16bf out;
#pragma unroll
  for (int i = 0; i < 8; ++i) { out[i] = lo[i]; out[i + 8] = hi[i]; }
  return out;
}

// ---------------------------------------------------------------------------
// TDM: load h_prev [64 rows x 1024 bf16, row stride 2048B] from global into the
// blocked LDS layout. 3D tile in 8B units: x=8 (64B run), y=64 rows (stride 256),
// z=32 k-blocks (stride 8). LDS pad: +16B per 256B stored (one pad per 4 rows).
// amdgpu-toolchain 6-arg builtin: (u32x4, i32x8, i32x4, i32x4, i32x8, i32 cpol)
// ---------------------------------------------------------------------------
__device__ __forceinline__ void tdm_load_h_blocked(unsigned lds_off,
                                                   unsigned long long gaddr) {
  tdm_u32x4 g0;
  g0[0] = 1u;                                      // count=1, user descriptor
  g0[1] = lds_off;                                 // lds_addr
  g0[2] = (unsigned)(gaddr & 0xFFFFFFFFull);       // global_addr low
  g0[3] = (unsigned)((gaddr >> 32) & 0x1FFFFFFull) | (2u << 30); // hi | type=2
  tdm_i32x8 g1;
  g1[0] = (int)((3u << 16)      // data_size = 8B
              | (1u << 20)      // pad_enable
              | (5u << 22)      // pad_interval: 64 DWORDs = 256B
              | (3u << 25));    // pad_amount:   4 DWORDs = 16B
  g1[1] = (int)(8u << 16);      // tensor_dim0 = 8 (bits 63:48)
  g1[2] = (int)(64u << 16);     // tensor_dim0 hi=0 | tensor_dim1 = 64 (bits 95:80)
  g1[3] = (int)(8u << 16);      // tensor_dim1 hi=0 | tile_dim0 = 8 (bits 127:112)
  g1[4] = (int)(64u | (32u << 16)); // tile_dim1 = 64, tile_dim2 = 32
  g1[5] = 256;                  // tensor_dim0_stride = 256 (2048B row)
  g1[6] = (int)(8u << 16);      // dim0_stride hi=0 | tensor_dim1_stride = 8 (64B)
  g1[7] = 0;
  tdm_i32x4 g2;
  g2[0] = 32;                   // tensor_dim2 = 32
  g2[1] = 0; g2[2] = 0; g2[3] = 0;
  tdm_i32x4 g3;
  g3[0] = 0; g3[1] = 0; g3[2] = 0; g3[3] = 0;
  tdm_i32x8 gz8;
#pragma unroll
  for (int i = 0; i < 8; ++i) gz8[i] = 0;
  __builtin_amdgcn_tensor_load_to_lds(g0, g1, g2, g3, gz8, 0);
}

// ---------------------------------------------------------------------------
// Grid-wide barrier (persistent kernel). Monotonic arrival counter in ws.
// ---------------------------------------------------------------------------
__device__ __forceinline__ void grid_sync(unsigned int* bar, unsigned nblk, int step) {
  __threadfence();
  __syncthreads();
  if (threadIdx.x == 0) {
    atomicAdd(&bar[0], 1u);
    unsigned target = nblk * (unsigned)(step + 1);
    volatile unsigned int* vb = bar;
    while (vb[0] < target) { __builtin_amdgcn_s_sleep(1); }
  }
  __syncthreads();
}

// ---------------------------------------------------------------------------
// Small prep kernels
// ---------------------------------------------------------------------------
__global__ void init_kernel(__bf16* hbuf0, unsigned int* bar) {
  int i = blockIdx.x * blockDim.x + threadIdx.x;
  if (i < BATCH * HID) hbuf0[i] = (__bf16)0.0f;
  if (i < 4) bar[i] = 0u;
}

__global__ void f32_to_bf16_kernel(const float* __restrict__ src,
                                   __bf16* __restrict__ dst, int n) {
  int i = blockIdx.x * blockDim.x + threadIdx.x;
  if (i < n) dst[i] = (__bf16)src[i];
}

__global__ void bias_sum_kernel(const float* __restrict__ a,
                                const float* __restrict__ b,
                                float* __restrict__ dst, int n) {
  int i = blockIdx.x * blockDim.x + threadIdx.x;
  if (i < n) dst[i] = a[i] + b[i];
}

// Gather embeddings, time-major rows (row = s*64 + b), convert to bf16.
__global__ void embed_kernel(const int* __restrict__ x,
                             const float* __restrict__ emb,
                             __bf16* __restrict__ X) {
  int row = blockIdx.x;               // 0..32767
  int s = row >> 6, b = row & 63;
  int tok = x[b * SEQ + s];
  const float* src = emb + (size_t)tok * EMB;
  __bf16* dst = X + (size_t)row * EMB;
  for (int i = threadIdx.x; i < EMB; i += blockDim.x) dst[i] = (__bf16)src[i];
}

// ---------------------------------------------------------------------------
// bf16 GEMM:  C[M,N] = bf16( A[M,K] * B[K,N] + bias[N] ), fp32 accumulate.
// Block: 256 threads (8 waves), tile 128x64, BK=32. Waves 4x2, each 32x32.
// LDS tiles use the padded-blocked layout (conflict-free frag loads).
// ---------------------------------------------------------------------------
__global__ void __launch_bounds__(256) gemm_bf16_kernel(
    const __bf16* __restrict__ A, const __bf16* __restrict__ Bw,
    const float* __restrict__ bias, __bf16* __restrict__ C,
    int M, int N, int K) {
  __shared__ __bf16 As[2 * BLK_STRIDE64];   // 128 rows (two 64-row blocks)
  __shared__ __bf16 BsT[BLK_STRIDE64];      // 64 cols transposed
  const int tid  = threadIdx.x;
  const int lane = tid & 31;
  const int w    = tid >> 5;
  const int wm   = w & 3;             // 0..3  (32-row strip)
  const int wn   = w >> 2;            // 0..1  (32-col strip)
  const int m0   = blockIdx.y * 128;
  const int n0   = blockIdx.x * 64;

  v8f acc[2][2];
#pragma unroll
  for (int a = 0; a < 2; ++a)
#pragma unroll
    for (int b = 0; b < 2; ++b)
#pragma unroll
      for (int i = 0; i < 8; ++i) acc[a][b][i] = 0.0f;

  for (int k0 = 0; k0 < K; k0 += 32) {
    {   // A tile 128x32 -> LDS blocked (each thread: one 32B chunk)
      int row = tid >> 1;             // 0..127
      int kc  = (tid & 1) * 16;
      const bf16x8* src = (const bf16x8*)(A + (size_t)(m0 + row) * K + k0 + kc);
      __bf16* dp = As + (row >> 6) * BLK_STRIDE64 + BLK_ROW(row & 63) + kc;
      ((bf16x8*)dp)[0] = src[0];
      ((bf16x8*)dp)[1] = src[1];
      if (k0 + 32 < K) __builtin_prefetch(src + 4, 0, 1);   // global_prefetch
    }
    {   // B tile 32x64 -> LDS transposed blocked
      int kk = tid >> 3;              // 0..31
      int nn = (tid & 7) * 8;
      bf16x8 v = *(const bf16x8*)(Bw + (size_t)(k0 + kk) * N + n0 + nn);
#pragma unroll
      for (int i = 0; i < 8; ++i) BsT[BLK_ROW(nn + i) + kk] = v[i];
    }
    __syncthreads();
    int ab = (wm >> 1);               // which 64-row block
    int r0 = (wm & 1) * 32;           // row inside block
    v16bf a0 = load_frag_a_blk(As + ab * BLK_STRIDE64, r0 + 0, lane);
    v16bf a1 = load_frag_a_blk(As + ab * BLK_STRIDE64, r0 + 16, lane);
    v16bf b0 = load_frag_b_blk(BsT, wn * 32 + 0, lane);
    v16bf b1 = load_frag_b_blk(BsT, wn * 32 + 16, lane);
    acc[0][0] = wmma_bf16(a0, b0, acc[0][0]);
    acc[0][1] = wmma_bf16(a0, b1, acc[0][1]);
    acc[1][0] = wmma_bf16(a1, b0, acc[1][0]);
    acc[1][1] = wmma_bf16(a1, b1, acc[1][1]);
    __syncthreads();
  }

  // Epilogue: bias add, convert, store. C/D layout: elem r -> row (lane>>4)*8+r.
  int nl = lane & 15, rb = (lane >> 4) * 8;
#pragma unroll
  for (int mt = 0; mt < 2; ++mt)
#pragma unroll
    for (int nt = 0; nt < 2; ++nt) {
      int col = n0 + wn * 32 + nt * 16 + nl;
      float bv = bias ? bias[col] : 0.0f;
#pragma unroll
      for (int r = 0; r < 8; ++r) {
        int row = m0 + wm * 32 + mt * 16 + rb + r;
        C[(size_t)row * N + col] = (__bf16)(acc[mt][nt][r] + bv);
      }
    }
}

// ---------------------------------------------------------------------------
// Persistent LSTM scan. 64 WGs x 256 threads; WG owns 16 hidden units
// (-> 64 gate columns: {c0..c0+15} + {1024,2048,3072} offsets).
// LDS (padded-blocked): WhhT 136K + Hs 136K + Gs 16K = 288K (< 320K WGP)
// ---------------------------------------------------------------------------
#define SCAN_NWG 64
#define SCAN_BLK_BYTES (BLK_STRIDE64 * 2)            // 4352 B per k-block
#define LDS_WHHT 0
#define LDS_HS   (32 * SCAN_BLK_BYTES)               // 139264
#define LDS_GS   (2 * 32 * SCAN_BLK_BYTES)           // 278528
#define SCAN_LDS (LDS_GS + BATCH * 64 * 4)           // 294912

__global__ void __launch_bounds__(256, 1) lstm_scan_kernel(
    const __bf16* __restrict__ gx,    // [SEQ*64, 4096] bf16, time-major
    const __bf16* __restrict__ whh,   // [1024, 4096] bf16
    __bf16* __restrict__ hbuf,        // [2][64][1024] bf16 (double buffered)
    unsigned int* __restrict__ bar) {
  extern __shared__ char smem[];
  __bf16* WhhT = (__bf16*)(smem + LDS_WHHT);   // blocked: [k/32][col][32]+pad
  __bf16* Hs   = (__bf16*)(smem + LDS_HS);     // blocked: [k/32][b][32]+pad
  float*  Gs   = (float*)(smem + LDS_GS);      // [b 0..63][col 0..63]

  const int tid  = threadIdx.x;
  const int lane = tid & 31;
  const int w    = tid >> 5;          // wave 0..7
  const int mi   = w & 3;             // batch 16-row tile
  const int gb   = (w >> 2) * 2;      // gate pair base: {0,1} or {2,3}
  const int wg   = blockIdx.x;        // 0..63
  const int c0   = wg * 16;           // hidden-unit chunk base

  // Stage this WG's w_hh slice into blocked LDS (one-time; reused 512x).
  for (int idx = tid; idx < 64 * HID; idx += 256) {
    int k = idx >> 6, col = idx & 63;
    int gcol = (col >> 4) * HID + c0 + (col & 15);
    WhhT[(k >> 5) * BLK_STRIDE64 + BLK_ROW(col) + (k & 31)] =
        whh[(size_t)k * G4 + gcol];
  }
  __syncthreads();

  float creg[4] = {0.f, 0.f, 0.f, 0.f};   // c-state: 4 (b,j) elems per thread

  for (int t = 0; t < SEQ; ++t) {
    // --- Stage h_prev (128 KB) via Tensor Data Mover into blocked layout ---
    const __bf16* hsrc = hbuf + (size_t)(t & 1) * BATCH * HID;
    if (w == 0) {
      tdm_load_h_blocked((unsigned)(size_t)Hs,
                         (unsigned long long)(size_t)hsrc);
      __builtin_amdgcn_s_wait_tensorcnt(0);
    }
    // --- Preload gates_x slice into Gs (fp32) ---
    for (int idx = tid; idx < BATCH * 64; idx += 256) {
      int b = idx >> 6, col = idx & 63;
      int gcol = (col >> 4) * HID + c0 + (col & 15);
      Gs[idx] = (float)gx[((size_t)t * BATCH + b) * G4 + gcol];
    }
    __syncthreads();

    // --- gates += h_prev @ Whh_slice : M=64,N=64,K=1024 via WMMA ---
    v8f acc0, acc1;
#pragma unroll
    for (int i = 0; i < 8; ++i) { acc0[i] = 0.f; acc1[i] = 0.f; }
#pragma unroll 8
    for (int kk = 0; kk < HID; kk += 32) {
      const __bf16* hb = Hs   + (kk >> 5) * BLK_STRIDE64;
      const __bf16* wb = WhhT + (kk >> 5) * BLK_STRIDE64;
      v16bf a  = load_frag_a_blk(hb, mi * 16, lane);
      v16bf b0 = load_frag_b_blk(wb, (gb + 0) * 16, lane);
      v16bf b1 = load_frag_b_blk(wb, (gb + 1) * 16, lane);
      acc0 = wmma_bf16(a, b0, acc0);
      acc1 = wmma_bf16(a, b1, acc1);
    }
    {   // accumulate into Gs (each element owned by exactly one lane)
      int n = lane & 15, rb = (lane >> 4) * 8;
#pragma unroll
      for (int r = 0; r < 8; ++r) {
        int b = mi * 16 + rb + r;
        Gs[b * 64 + (gb + 0) * 16 + n] += acc0[r];
        Gs[b * 64 + (gb + 1) * 16 + n] += acc1[r];
      }
    }
    __syncthreads();

    // --- Activations + state update: 4 (b,j) elements per thread ---
    __bf16* hdst = hbuf + (size_t)((t + 1) & 1) * BATCH * HID;
#pragma unroll
    for (int e = 0; e < 4; ++e) {
      int el = tid * 4 + e;           // 0..1023 = b*16 + j
      int b = el >> 4, j = el & 15;
      float gi = Gs[b * 64 + 0 * 16 + j];
      float gf = Gs[b * 64 + 1 * 16 + j];
      float gg = Gs[b * 64 + 2 * 16 + j];
      float go = Gs[b * 64 + 3 * 16 + j];
      float i_ = 1.0f / (1.0f + __expf(-gi));
      float f_ = 1.0f / (1.0f + __expf(-gf));
      float g_ = tanhf(gg);
      float o_ = 1.0f / (1.0f + __expf(-go));
      float c  = f_ * creg[e] + i_ * g_;
      creg[e] = c;
      hdst[b * HID + c0 + j] = (__bf16)(o_ * tanhf(c));
    }
    grid_sync(bar, SCAN_NWG, t);
  }
}

// ---------------------------------------------------------------------------
// Classifier: out[64,2] = h_last @ out_w + out_b  (tiny, fp32)
// ---------------------------------------------------------------------------
__global__ void out_kernel(const __bf16* __restrict__ h,
                           const float* __restrict__ W,
                           const float* __restrict__ bv,
                           float* __restrict__ out) {
  int tid = threadIdx.x;
  if (tid < 128) {
    int b = tid >> 1, c = tid & 1;
    float s = bv[c];
    for (int k = 0; k < HID; ++k) s += (float)h[b * HID + k] * W[k * 2 + c];
    out[b * 2 + c] = s;
  }
}

// ---------------------------------------------------------------------------
// Workspace layout (all 256-B aligned)
// ---------------------------------------------------------------------------
static const size_t OFF_BAR  = 0;
static const size_t OFF_HBUF = 256;
static const size_t OFF_BW   = OFF_HBUF + (size_t)2 * BATCH * HID * 2;
static const size_t OFF_WIH  = OFF_BW   + (size_t)EMB * EMB * 2;
static const size_t OFF_WHH  = OFF_WIH  + (size_t)EMB * G4 * 2;
static const size_t OFF_BSUM = OFF_WHH  + (size_t)HID * G4 * 2;
static const size_t OFF_X    = OFF_BSUM + (size_t)G4 * 4;
static const size_t OFF_HIN  = OFF_X    + (size_t)ROWS * EMB * 2;
static const size_t OFF_GX   = OFF_HIN  + (size_t)ROWS * EMB * 2;

extern "C" void kernel_launch(void* const* d_in, const int* in_sizes, int n_in,
                              void* d_out, int out_size, void* d_ws, size_t ws_size,
                              hipStream_t stream) {
  const int*   x        = (const int*)  d_in[0];
  const float* emb      = (const float*)d_in[1];
  const float* bridge_w = (const float*)d_in[2];
  const float* bridge_b = (const float*)d_in[3];
  const float* w_ih     = (const float*)d_in[4];
  const float* w_hh     = (const float*)d_in[5];
  const float* b_ih     = (const float*)d_in[6];
  const float* b_hh     = (const float*)d_in[7];
  const float* out_w    = (const float*)d_in[8];
  const float* out_b    = (const float*)d_in[9];
  float* out = (float*)d_out;

  char* ws = (char*)d_ws;
  unsigned int* bar = (unsigned int*)(ws + OFF_BAR);
  __bf16* hbuf = (__bf16*)(ws + OFF_HBUF);
  __bf16* bwb  = (__bf16*)(ws + OFF_BW);
  __bf16* wihb = (__bf16*)(ws + OFF_WIH);
  __bf16* whhb = (__bf16*)(ws + OFF_WHH);
  float*  bsum = (float*) (ws + OFF_BSUM);
  __bf16* X    = (__bf16*)(ws + OFF_X);
  __bf16* Hin  = (__bf16*)(ws + OFF_HIN);
  __bf16* Gx   = (__bf16*)(ws + OFF_GX);

  // 0) re-init barrier + h0 (harness poisons ws once; we must self-init)
  init_kernel<<<(BATCH * HID + 255) / 256, 256, 0, stream>>>(hbuf, bar);
  // 1) weight conversions to bf16
  f32_to_bf16_kernel<<<(EMB * EMB + 255) / 256, 256, 0, stream>>>(bridge_w, bwb, EMB * EMB);
  f32_to_bf16_kernel<<<(EMB * G4 + 255) / 256, 256, 0, stream>>>(w_ih, wihb, EMB * G4);
  f32_to_bf16_kernel<<<(HID * G4 + 255) / 256, 256, 0, stream>>>(w_hh, whhb, HID * G4);
  bias_sum_kernel<<<(G4 + 255) / 256, 256, 0, stream>>>(b_ih, b_hh, bsum, G4);
  // 2) embedding gather (time-major) -> X bf16
  embed_kernel<<<ROWS, 256, 0, stream>>>(x, emb, X);
  // 3) bridge GEMM: Hin = X @ bridge_w + bridge_b   [32768,512]x[512,512]
  gemm_bf16_kernel<<<dim3(EMB / 64, ROWS / 128), 256, 0, stream>>>(
      X, bwb, bridge_b, Hin, ROWS, EMB, EMB);
  // 4) input-gate GEMM: Gx = Hin @ w_ih + (b_ih+b_hh)  [32768,512]x[512,4096]
  gemm_bf16_kernel<<<dim3(G4 / 64, ROWS / 128), 256, 0, stream>>>(
      Hin, wihb, bsum, Gx, ROWS, G4, EMB);
  // 5) persistent LSTM scan (64 WGs, grid barrier per step, w_hh in LDS, TDM h stage)
  lstm_scan_kernel<<<SCAN_NWG, 256, SCAN_LDS, stream>>>(Gx, whhb, hbuf, bar);
  // 6) classifier head (h_last is in hbuf[0] after 512 steps)
  out_kernel<<<1, 128, 0, stream>>>(hbuf, out_w, out_b, out);
}
